// MLA_50946902065660
// MI455X (gfx1250) — matrix operand order repaired and checked by reading, not compiled, not run on hardware
//
#include <hip/hip_runtime.h>

// ---------------------------------------------------------------------------
// MLA (DeepSeek-style) forward, MI455X / gfx1250, bf16 WMMA + fp32 accumulate
// B=2 S=2048 D=2048 H=16 NOPE=128 ROPE=64 V=128 C=512
// ---------------------------------------------------------------------------

typedef __attribute__((ext_vector_type(16))) __bf16 bf16x16;
typedef __attribute__((ext_vector_type(8)))  float  f32x8;
typedef int gint4 __attribute__((vector_size(16)));   // matches builtin param type

#if defined(__AMDGCN__) && __has_builtin(__builtin_amdgcn_global_load_async_to_lds_b128)
#define USE_ASYNC_LDS 1
#define ASYNC_CP16(gsrc, ldst)                                                \
    __builtin_amdgcn_global_load_async_to_lds_b128(                           \
        (__attribute__((address_space(1))) gint4*)(gsrc),                     \
        (__attribute__((address_space(3))) gint4*)(ldst), 0, 0)
#define ASYNC_WAIT() asm volatile("s_wait_asynccnt 0x0" ::: "memory")
#else
#define USE_ASYNC_LDS 0
#endif

__device__ __forceinline__ unsigned short f2bf(float f) {
    union { float f; unsigned u; } v; v.f = f;
    unsigned r = v.u + 0x7FFFu + ((v.u >> 16) & 1u);   // round-nearest-even
    return (unsigned short)(r >> 16);
}

__device__ __forceinline__ f32x8 zero8() {
    f32x8 z = {0.f,0.f,0.f,0.f,0.f,0.f,0.f,0.f}; return z;
}

// Build a 16-element bf16 fragment from two 16B chunks.
__device__ __forceinline__ bf16x16 ldfrag(const unsigned short* p0,
                                          const unsigned short* p1) {
    union { struct { uint4 a, b; } u; bf16x16 v; } f;
    f.u.a = *(const uint4*)p0;
    f.u.b = *(const uint4*)p1;
    return f.v;
}

__device__ __forceinline__ f32x8 wmma_bf16(bf16x16 a, bf16x16 b, f32x8 c) {
    return __builtin_amdgcn_wmma_f32_16x16x32_bf16(
        false, a, false, b, (short)0, c, false, false);
}

// ---------------------------------------------------------------------------
// Generic batched GEMM:  C[m][n] = sum_k A[m][k] * W[n][k]   (A,W bf16 bits)
// Block tile 256x128, K-chunk 32, 8 waves (4 x 2), wave tile 64x64 (16 frags).
// M must be a multiple of 256 (true for all call sites); N may be ragged
// (handled by clamping W row addresses + epilogue guard: WMMA column n of D
// depends only on column n of B, so clamped-garbage columns are never stored).
// Async global->LDS staging with ping-pong buffers when available.
// ---------------------------------------------------------------------------
#define GPAD 40   // LDS row stride in ushorts (32 data + 8 pad)

__global__ __launch_bounds__(256) void gemm_bf16_kernel(
    const unsigned short* __restrict__ A, const unsigned short* __restrict__ W,
    void* __restrict__ Cout, int M, int N, int K,
    long long sA, long long sW, long long sC, int ldc, int out_bf16)
{
#if USE_ASYNC_LDS
    __shared__ unsigned short As[2][256 * GPAD];
    __shared__ unsigned short Ws[2][128 * GPAD];
#else
    __shared__ unsigned short As[1][256 * GPAD];
    __shared__ unsigned short Ws[1][128 * GPAD];
#endif

    const int tid  = threadIdx.x;
    const int wave = tid >> 5, lane = tid & 31;
    const int l16 = lane & 15, lhi = lane >> 4;
    const int wm = wave & 3, wn = wave >> 2;          // 4 waves (M) x 2 waves (N)
    const int row0 = blockIdx.y * 256;
    const int col0 = blockIdx.x * 128;
    const long long batch = blockIdx.z;

    const unsigned short* Ab = A + batch * sA;
    const unsigned short* Wb = W + batch * sW;

    f32x8 acc[4][4];
    #pragma unroll
    for (int i = 0; i < 4; ++i)
        #pragma unroll
        for (int j = 0; j < 4; ++j) acc[i][j] = zero8();

    // per-thread staging coordinates (A: 4 chunks, W: 2 chunks of 16B)
    const int srow = tid >> 2, spart = tid & 3;

    auto compute = [&](const unsigned short* as, const unsigned short* ws) {
        bf16x16 bfr[4];
        #pragma unroll
        for (int j = 0; j < 4; ++j) {
            const unsigned short* p = &ws[(wn * 64 + j * 16 + l16) * GPAD + lhi * 16];
            bfr[j] = ldfrag(p, p + 8);
        }
        #pragma unroll
        for (int i = 0; i < 4; ++i) {
            const unsigned short* p = &as[(wm * 64 + i * 16 + l16) * GPAD + lhi * 16];
            bf16x16 afr = ldfrag(p, p + 8);
            #pragma unroll
            for (int j = 0; j < 4; ++j)
                acc[i][j] = wmma_bf16(afr, bfr[j], acc[i][j]);
        }
    };

#if USE_ASYNC_LDS
    auto issue = [&](int bf, int kk0) {
        #pragma unroll
        for (int k = 0; k < 4; ++k) {
            int row = srow + k * 64;
            ASYNC_CP16(Ab + (long long)(row0 + row) * K + kk0 + spart * 8,
                       &As[bf][row * GPAD + spart * 8]);
        }
        #pragma unroll
        for (int k = 0; k < 2; ++k) {
            int row = srow + k * 64;
            int wr = col0 + row; if (wr >= N) wr = N - 1;   // clamp, see header note
            ASYNC_CP16(Wb + (long long)wr * K + kk0 + spart * 8,
                       &Ws[bf][row * GPAD + spart * 8]);
        }
    };

    issue(0, 0);
    ASYNC_WAIT();
    __syncthreads();
    int buf = 0;
    for (int kk0 = 0; kk0 < K; kk0 += 32) {
        if (kk0 + 32 < K) issue(buf ^ 1, kk0 + 32);     // stage next while computing
        compute(&As[buf][0], &Ws[buf][0]);
        ASYNC_WAIT();
        __syncthreads();
        buf ^= 1;
    }
#else
    for (int kk0 = 0; kk0 < K; kk0 += 32) {
        #pragma unroll
        for (int k = 0; k < 4; ++k) {
            int row = srow + k * 64;
            *(uint4*)&As[0][row * GPAD + spart * 8] =
                *(const uint4*)(Ab + (long long)(row0 + row) * K + kk0 + spart * 8);
        }
        #pragma unroll
        for (int k = 0; k < 2; ++k) {
            int row = srow + k * 64;
            int wr = col0 + row; if (wr >= N) wr = N - 1;
            *(uint4*)&Ws[0][row * GPAD + spart * 8] =
                *(const uint4*)(Wb + (long long)wr * K + kk0 + spart * 8);
        }
        __syncthreads();
        compute(&As[0][0], &Ws[0][0]);
        __syncthreads();
    }
#endif

    // epilogue: C fragment layout — lane n = N column, VGPR r -> M = r + 8*lhi
    #pragma unroll
    for (int i = 0; i < 4; ++i)
        #pragma unroll
        for (int j = 0; j < 4; ++j) {
            int gmBase = row0 + wm * 64 + i * 16 + lhi * 8;
            int gn     = col0 + wn * 64 + j * 16 + l16;
            if (gn < N) {
                #pragma unroll
                for (int r = 0; r < 8; ++r) {
                    long long idx = batch * sC + (long long)(gmBase + r) * ldc + gn;
                    float v = acc[i][j][r];
                    if (out_bf16) ((unsigned short*)Cout)[idx] = f2bf(v);
                    else          ((float*)Cout)[idx] = v;
                }
            }
        }
}

// ---------------------------------------------------------------------------
// Elementwise / prep kernels
// ---------------------------------------------------------------------------
__global__ void f2bf_kernel(const float* __restrict__ in,
                            unsigned short* __restrict__ out, long long n) {
    long long i = (long long)blockIdx.x * 256 + threadIdx.x;
    long long stride = (long long)gridDim.x * 256;
    for (; i < n; i += stride) out[i] = f2bf(in[i]);
}

// w2a[h][c][d] = wkv_b[(h*256+d)*512 + c]   (absorb-q weight, [N=512][K=128])
__global__ void prep_w2a_kernel(const float* __restrict__ wkvb,
                                unsigned short* __restrict__ w2a) {
    long long i = (long long)blockIdx.x * 256 + threadIdx.x;
    if (i >= 16LL * 512 * 128) return;
    int d = (int)(i & 127);
    long long rest = i >> 7;
    int c = (int)(rest & 511);
    int h = (int)(rest >> 9);
    w2a[i] = f2bf(wkvb[((long long)(h * 256 + d)) * 512 + c]);
}

// wbv[h][v][c] = wkv_b[(h*256+128+v)*512 + c]   ([N=128][K=512])
__global__ void prep_wbv_kernel(const float* __restrict__ wkvb,
                                unsigned short* __restrict__ wbv) {
    long long i = (long long)blockIdx.x * 256 + threadIdx.x;
    if (i >= 16LL * 128 * 512) return;
    int c = (int)(i & 511);
    long long rest = i >> 9;
    int vv = (int)(rest & 127);
    int h  = (int)(rest >> 7);
    wbv[i] = f2bf(wkvb[((long long)(h * 256 + 128 + vv)) * 512 + c]);
}

// Split q: nope part -> qa_in[h][m][128] (bf16), rope(q_pe) -> Qcat[h][m][512..575]
__global__ __launch_bounds__(64) void rope_q_kernel(
    const float* __restrict__ q_tmp, const float* __restrict__ cosv,
    const float* __restrict__ sinv,
    unsigned short* __restrict__ qa_in, unsigned short* __restrict__ Qcat)
{
    int id = blockIdx.x;          // m*16 + h, m = b*2048+s
    int m = id >> 4, h = id & 15;
    int s = m & 2047;
    const float* qr = q_tmp + (long long)m * 3072 + h * 192;
    unsigned short* qa = qa_in + ((long long)(h * 4096 + m)) * 128;
    unsigned short* qc = Qcat  + ((long long)(h * 4096 + m)) * 576 + 512;
    int t = threadIdx.x;
    for (int i = t; i < 128; i += 64) qa[i] = f2bf(qr[i]);
    if (t < 32) {
        float xr = qr[128 + 2 * t], xi = qr[128 + 2 * t + 1];
        float c = cosv[s * 32 + t], sn = sinv[s * 32 + t];
        qc[2 * t]     = f2bf(xr * c - xi * sn);
        qc[2 * t + 1] = f2bf(xr * sn + xi * c);
    }
}

// RMSNorm(kv) -> Kcat[m][0..511] and transposed Vt[b][c][t]; rope(k_pe) -> Kcat[m][512..575]
__global__ __launch_bounds__(128) void kv_kernel(
    const float* __restrict__ kvr, const float* __restrict__ w,
    const float* __restrict__ cosv, const float* __restrict__ sinv,
    unsigned short* __restrict__ Kcat, unsigned short* __restrict__ Vt)
{
    __shared__ float red[128];
    __shared__ float rs;
    int m = blockIdx.x;                 // b*2048 + t
    int b = m >> 11, ti = m & 2047;
    const float* row = kvr + (long long)m * 576;
    int t = threadIdx.x;
    float v[4], ss = 0.f;
    for (int j = 0; j < 4; ++j) { v[j] = row[t + j * 128]; ss += v[j] * v[j]; }
    red[t] = ss; __syncthreads();
    for (int st = 64; st > 0; st >>= 1) {
        if (t < st) red[t] += red[t + st];
        __syncthreads();
    }
    if (t == 0) rs = rsqrtf(red[0] / 512.f + 1e-6f);
    __syncthreads();
    float r = rs;
    for (int j = 0; j < 4; ++j) {
        int c = t + j * 128;
        unsigned short bf = f2bf(v[j] * r * w[c]);
        Kcat[(long long)m * 576 + c] = bf;
        Vt[((long long)b * 512 + c) * 2048 + ti] = bf;
    }
    if (t < 32) {
        float xr = row[512 + 2 * t], xi = row[512 + 2 * t + 1];
        float c = cosv[ti * 32 + t], sn = sinv[ti * 32 + t];
        Kcat[(long long)m * 576 + 512 + 2 * t]     = f2bf(xr * c - xi * sn);
        Kcat[(long long)m * 576 + 512 + 2 * t + 1] = f2bf(xr * sn + xi * c);
    }
}

// ---------------------------------------------------------------------------
// Flash attention: per (b,h), 32-query-row tile; key tiles of 64; d_k=576, d_v=512
// 8 waves: each computes one 16x16 score fragment; O (32x512 fp32) distributed
// 8 fragments per wave. Causal mask applied analytically. Softmax parallel 8x/row.
// ---------------------------------------------------------------------------
#define QPAD 584
#define PPAD 72

__global__ __launch_bounds__(256) void attn_kernel(
    const unsigned short* __restrict__ Qcat,  // [H][4096][576]
    const unsigned short* __restrict__ Kcat,  // [4096][576]
    const unsigned short* __restrict__ Vt,    // [B][512][2048]
    unsigned short* __restrict__ Obuf)        // [H][4096][512]
{
    __shared__ unsigned short Qs[32 * QPAD];
    __shared__ float Ps[32 * 64];
    __shared__ unsigned short Pb[32 * PPAD];
    __shared__ float m_s[32], l_s[32], a_s[32], mn_s[32];
    __shared__ float red2[256];

    const int qt = blockIdx.x, h = blockIdx.y, b = blockIdx.z;
    const int s0 = qt * 32;
    const int tid = threadIdx.x, wave = tid >> 5, lane = tid & 31;
    const int l16 = lane & 15, lhi = lane >> 4;

    const unsigned short* Qg = Qcat + ((long long)(h * 4096 + b * 2048 + s0)) * 576;
    const unsigned short* Kg = Kcat + ((long long)(b * 2048)) * 576;
    const unsigned short* Vg = Vt + (long long)b * 512 * 2048;

    // stage Q tile (32 x 576) into LDS
#if USE_ASYNC_LDS
    for (int c = tid; c < 32 * 72; c += 256) {
        int row = c / 72, part = c % 72;
        ASYNC_CP16(Qg + (long long)row * 576 + part * 8, &Qs[row * QPAD + part * 8]);
    }
    if (tid < 32) { m_s[tid] = -1e30f; l_s[tid] = 0.f; }
    ASYNC_WAIT();
#else
    for (int c = tid; c < 32 * 72; c += 256) {
        int row = c / 72, part = c % 72;
        *(uint4*)&Qs[row * QPAD + part * 8] =
            *(const uint4*)(Qg + (long long)row * 576 + part * 8);
    }
    if (tid < 32) { m_s[tid] = -1e30f; l_s[tid] = 0.f; }
#endif
    __syncthreads();

    const int snt = wave & 3;                    // score fragment coords
    const int smt = wave >> 2;
    const int omt = wave >> 2, ocg = wave & 3;   // O fragment coords

    f32x8 Oacc[8];
    #pragma unroll
    for (int j = 0; j < 8; ++j) Oacc[j] = zero8();

    const float scale = 0.07216878364870323f;    // 1/sqrt(192)
    const int ktmax = (s0 + 31) >> 6;

    for (int kt = 0; kt <= ktmax; ++kt) {
        const int t0 = kt * 64;

        // prefetch next key tile's K rows and V columns into cache
        if (kt < ktmax) {
            const unsigned short* nK = Kg + (long long)(t0 + 64) * 576;
            #pragma unroll
            for (int k = 0; k < 2; ++k) {
                int c = tid + k * 256;                 // 512 segments of 144B
                __builtin_prefetch(nK + (long long)(c >> 3) * 576 + (c & 7) * 72, 0, 3);
            }
            #pragma unroll
            for (int k = 0; k < 2; ++k) {
                int c = tid + k * 256;                 // 512 V rows, 128B each
                __builtin_prefetch(Vg + (long long)c * 2048 + t0 + 64, 0, 3);
            }
        }

        // ---- scores: 16x16 fragment, K-dim 576 = 18 WMMAs
        f32x8 sc = zero8();
        const int qrow = smt * 16 + l16;
        const unsigned short* Kr = Kg + (long long)(t0 + snt * 16 + l16) * 576;
        #pragma unroll 6
        for (int kk = 0; kk < 18; ++kk) {
            const unsigned short* pa = &Qs[qrow * QPAD + kk * 32 + lhi * 16];
            bf16x16 afr = ldfrag(pa, pa + 8);
            const unsigned short* pb = Kr + kk * 32 + lhi * 16;
            bf16x16 bfr = ldfrag(pb, pb + 8);
            sc = wmma_bf16(afr, bfr, sc);
        }
        // scale + causal mask, spill to LDS
        #pragma unroll
        for (int r = 0; r < 8; ++r) {
            int mm = smt * 16 + lhi * 8 + r;
            int tt = t0 + snt * 16 + l16;
            float v = sc[r] * scale;
            if (tt > s0 + mm) v = -1e38f;
            Ps[mm * 64 + snt * 16 + l16] = v;
        }
        __syncthreads();

        // ---- online softmax, parallelized 8 threads per query row
        {
            int r = tid >> 3, sl = tid & 7;
            float mx = -3e38f;
            #pragma unroll
            for (int j = 0; j < 8; ++j) mx = fmaxf(mx, Ps[r * 64 + sl * 8 + j]);
            red2[tid] = mx;
        }
        __syncthreads();
        if (tid < 32) {
            int r = tid;
            float mx = m_s[r];
            #pragma unroll
            for (int j = 0; j < 8; ++j) mx = fmaxf(mx, red2[r * 8 + j]);
            a_s[r]  = __expf(m_s[r] - mx);
            mn_s[r] = mx;
        }
        __syncthreads();
        {
            int r = tid >> 3, sl = tid & 7;
            float mnew = mn_s[r];
            float sum = 0.f;
            #pragma unroll
            for (int j = 0; j < 8; ++j) {
                float p = __expf(Ps[r * 64 + sl * 8 + j] - mnew);
                sum += p;
                Pb[r * PPAD + sl * 8 + j] = f2bf(p);
            }
            red2[tid] = sum;
        }
        __syncthreads();
        if (tid < 32) {
            int r = tid;
            float s = 0.f;
            #pragma unroll
            for (int j = 0; j < 8; ++j) s += red2[r * 8 + j];
            l_s[r] = l_s[r] * a_s[r] + s;
            m_s[r] = mn_s[r];
        }
        __syncthreads();

        // ---- rescale O, then O += P (32x64) x V (64x512)
        float arow[8];
        #pragma unroll
        for (int r = 0; r < 8; ++r) arow[r] = a_s[omt * 16 + lhi * 8 + r];
        #pragma unroll
        for (int j = 0; j < 8; ++j)
            #pragma unroll
            for (int r = 0; r < 8; ++r) Oacc[j][r] *= arow[r];

        #pragma unroll
        for (int j = 0; j < 8; ++j) {
            int ct = ocg * 8 + j;
            f32x8 o = Oacc[j];
            #pragma unroll
            for (int kk = 0; kk < 2; ++kk) {
                const unsigned short* pa =
                    &Pb[(omt * 16 + l16) * PPAD + kk * 32 + lhi * 16];
                bf16x16 afr = ldfrag(pa, pa + 8);
                const unsigned short* pv =
                    Vg + (long long)(ct * 16 + l16) * 2048 + t0 + kk * 32 + lhi * 16;
                bf16x16 bfr = ldfrag(pv, pv + 8);
                o = wmma_bf16(afr, bfr, o);
            }
            Oacc[j] = o;
        }
        __syncthreads();
    }

    // ---- normalize and write out (bf16)
    if (tid < 32) a_s[tid] = (l_s[tid] > 0.f) ? 1.f / l_s[tid] : 0.f;
    __syncthreads();
    float inv[8];
    #pragma unroll
    for (int r = 0; r < 8; ++r) inv[r] = a_s[omt * 16 + lhi * 8 + r];
    unsigned short* Og = Obuf + ((long long)(h * 4096 + b * 2048 + s0)) * 512;
    #pragma unroll
    for (int j = 0; j < 8; ++j) {
        int ct = ocg * 8 + j;
        #pragma unroll
        for (int r = 0; r < 8; ++r) {
            int mm = omt * 16 + lhi * 8 + r;
            Og[(long long)mm * 512 + ct * 16 + l16] = f2bf(Oacc[j][r] * inv[r]);
        }
    }
}

// ---------------------------------------------------------------------------
// Host launcher
// ---------------------------------------------------------------------------
extern "C" void kernel_launch(void* const* d_in, const int* in_sizes, int n_in,
                              void* d_out, int out_size, void* d_ws, size_t ws_size,
                              hipStream_t stream) {
    (void)in_sizes; (void)n_in; (void)out_size; (void)ws_size;

    const float* x     = (const float*)d_in[0];
    const float* cosv  = (const float*)d_in[1];
    const float* sinv  = (const float*)d_in[2];
    // d_in[3] = mask (causal; applied analytically)
    const float* wq    = (const float*)d_in[4];
    const float* wkva  = (const float*)d_in[5];
    const float* kvw   = (const float*)d_in[6];
    const float* wkvb  = (const float*)d_in[7];
    const float* wo    = (const float*)d_in[8];
    float* out = (float*)d_out;

    // workspace carve-up
    char* ws = (char*)d_ws;
    size_t off = 0;
    auto alloc = [&](size_t bytes) -> char* {
        char* p = ws + off;
        off = (off + bytes + 255) & ~(size_t)255;
        return p;
    };
    unsigned short* x_bf    = (unsigned short*)alloc(4096LL * 2048 * 2);
    unsigned short* wq_bf   = (unsigned short*)alloc(3072LL * 2048 * 2);
    unsigned short* wkva_bf = (unsigned short*)alloc(576LL  * 2048 * 2);
    unsigned short* wo_bf   = (unsigned short*)alloc(2048LL * 2048 * 2);
    unsigned short* w2a_bf  = (unsigned short*)alloc(16LL * 512 * 128 * 2);
    unsigned short* wbv_bf  = (unsigned short*)alloc(16LL * 128 * 512 * 2);
    float*          q_tmp   = (float*)alloc(4096LL * 3072 * 4);
    float*          kvr     = (float*)alloc(4096LL * 576 * 4);
    unsigned short* qa_in   = (unsigned short*)alloc(16LL * 4096 * 128 * 2);
    unsigned short* Qcat    = (unsigned short*)alloc(16LL * 4096 * 576 * 2);
    unsigned short* Kcat    = (unsigned short*)alloc(4096LL * 576 * 2);
    unsigned short* Vt      = (unsigned short*)alloc(2LL * 512 * 2048 * 2);
    unsigned short* o_buf   = (unsigned short*)alloc(16LL * 4096 * 512 * 2);
    unsigned short* o2      = (unsigned short*)alloc(4096LL * 2048 * 2);

    // 1) bf16 conversions
    f2bf_kernel<<<2048, 256, 0, stream>>>(x,    x_bf,    4096LL * 2048);
    f2bf_kernel<<<2048, 256, 0, stream>>>(wq,   wq_bf,   3072LL * 2048);
    f2bf_kernel<<<1024, 256, 0, stream>>>(wkva, wkva_bf, 576LL  * 2048);
    f2bf_kernel<<<2048, 256, 0, stream>>>(wo,   wo_bf,   2048LL * 2048);
    prep_w2a_kernel<<<(16 * 512 * 128 + 255) / 256, 256, 0, stream>>>(wkvb, w2a_bf);
    prep_wbv_kernel<<<(16 * 128 * 512 + 255) / 256, 256, 0, stream>>>(wkvb, wbv_bf);

    // 2) q = x @ wq^T   (4096 x 3072, K=2048) -> fp32
    gemm_bf16_kernel<<<dim3(3072 / 128, 4096 / 256, 1), 256, 0, stream>>>(
        x_bf, wq_bf, q_tmp, 4096, 3072, 2048, 0, 0, 0, 3072, 0);
    // 3) kvr = x @ wkv_a^T  (4096 x 576, K=2048) -> fp32
    gemm_bf16_kernel<<<dim3((576 + 127) / 128, 4096 / 256, 1), 256, 0, stream>>>(
        x_bf, wkva_bf, kvr, 4096, 576, 2048, 0, 0, 0, 576, 0);

    // 4) split/rope q; rmsnorm+rope kv
    rope_q_kernel<<<4096 * 16, 64, 0, stream>>>(q_tmp, cosv, sinv, qa_in, Qcat);
    kv_kernel<<<4096, 128, 0, stream>>>(kvr, kvw, cosv, sinv, Kcat, Vt);

    // 5) absorbed q: per-head (4096 x 512, K=128) -> bf16 into Qcat[:, :512]
    gemm_bf16_kernel<<<dim3(512 / 128, 4096 / 256, 16), 256, 0, stream>>>(
        qa_in, w2a_bf, Qcat, 4096, 512, 128,
        4096LL * 128, 512LL * 128, 4096LL * 576, 576, 1);

    // 6) flash attention
    attn_kernel<<<dim3(64, 16, 2), 256, 0, stream>>>(Qcat, Kcat, Vt, o_buf);

    // 7) v-projection: per-head (4096 x 128, K=512) -> bf16 into o2[:, h*128:]
    gemm_bf16_kernel<<<dim3(1, 4096 / 256, 16), 256, 0, stream>>>(
        o_buf, wbv_bf, o2, 4096, 128, 512,
        4096LL * 512, 128LL * 512, 128LL, 2048, 1);

    // 8) output projection: (4096 x 2048, K=2048) -> fp32 d_out
    gemm_bf16_kernel<<<dim3(2048 / 128, 4096 / 256, 1), 256, 0, stream>>>(
        o2, wo_bf, out, 4096, 2048, 2048, 0, 0, 0, 2048, 0);
}
